// Sweep_Mamba_84138409328980
// MI455X (gfx1250) — compile-verified, matrix operand order checked by reading, not run on hardware
//
#include <hip/hip_runtime.h>
#include <hip/hip_bf16.h>
#include <math.h>

typedef _Float16 h16;
typedef __attribute__((ext_vector_type(16))) _Float16 v16h;
typedef __attribute__((ext_vector_type(8)))  _Float16 v8h;
typedef __attribute__((ext_vector_type(8)))  float    v8f;

#define DEV_INLINE __device__ __forceinline__

DEV_INLINE float siluf(float x) { return x / (1.f + __expf(-x)); }
DEV_INLINE float softplusf(float x) { return (x > 20.f) ? x : log1pf(__expf(x)); }

static inline int round16(int v) { return (v + 15) & ~15; }
static inline int round32(int v) { return (v + 31) & ~31; }

// ---------------------------------------------------------------------------
// Input LayerNorm over DIM=1024 per (b,h,w) row; x layout (B,DIM,H,W).
// Writes f16 row-major (1536 x 1024) for WMMA GEMM.
// ---------------------------------------------------------------------------
__global__ __launch_bounds__(256)
void k_ln_in(const float* __restrict__ x, const float* __restrict__ g,
             const float* __restrict__ b, h16* __restrict__ out) {
    __shared__ float s1[256], s2[256];
    const int row = blockIdx.x;              // b*48 + l
    const int bb = row / 48, l = row % 48;
    const float* xr = x + (size_t)bb * 49152 + l;  // channel stride = 48
    float s = 0.f, q = 0.f;
    for (int c = threadIdx.x; c < 1024; c += 256) {
        float v = xr[(size_t)c * 48];
        s += v; q += v * v;
    }
    s1[threadIdx.x] = s; s2[threadIdx.x] = q; __syncthreads();
    for (int off = 128; off > 0; off >>= 1) {
        if ((int)threadIdx.x < off) { s1[threadIdx.x] += s1[threadIdx.x + off];
                                      s2[threadIdx.x] += s2[threadIdx.x + off]; }
        __syncthreads();
    }
    const float mean = s1[0] / 1024.f;
    const float var  = s2[0] / 1024.f - mean * mean;
    const float inv  = rsqrtf(var + 1e-5f);
    for (int c = threadIdx.x; c < 1024; c += 256) {
        float v = (xr[(size_t)c * 48] - mean) * inv * g[c] + b[c];
        out[(size_t)row * 1024 + c] = (h16)v;
    }
}

// ---------------------------------------------------------------------------
// Weight f32 -> zero-padded f16 (Npad x Kpad), row-major.
// ---------------------------------------------------------------------------
__global__ __launch_bounds__(256)
void k_w2h(const float* __restrict__ W, h16* __restrict__ out,
           int N, int K, int Npad, int Kpad) {
    int idx = blockIdx.x * 256 + threadIdx.x;
    if (idx >= Npad * Kpad) return;
    int n = idx / Kpad, k = idx % Kpad;
    out[idx] = (n < N && k < K) ? (h16)W[(size_t)n * K + k] : (h16)0.f;
}

// ---------------------------------------------------------------------------
// WMMA GEMM on padded f16 operands:
//   C(M,N) = A(Mpad,Kpad)[f16,lda=Kpad] @ B(Npad,Kpad)[f16,ldb=Kpad]^T + bias
// One wave per 16x16 tile.  Fragment = two contiguous b128 loads per operand:
// f16 A-layout per lane holds halves at [k0+grp*8, +8) and [k0+16+grp*8, +8).
// All bounds handled by zero padding -> branch-free inner loop.
// ---------------------------------------------------------------------------
__global__ __launch_bounds__(32)
void k_gemm(const h16* __restrict__ A, int lda,
            const h16* __restrict__ Bw, int ldb,
            const float* __restrict__ bias,
            float* __restrict__ C, int ldc, h16* __restrict__ C16,
            int N, int K) {
    __builtin_amdgcn_s_wait_tensorcnt(0);
    const int lane = threadIdx.x;
    const int grp = lane >> 4, lr = lane & 15;
    const int tM = blockIdx.x * 16, tN = blockIdx.y * 16;
    const h16* ap = A  + (size_t)(tM + lr) * lda + grp * 8;
    const h16* bp = Bw + (size_t)(tN + lr) * ldb + grp * 8;
    v8f acc = {};
#pragma unroll 2
    for (int k0 = 0; k0 < K; k0 += 32) {
        if (k0 + 32 < K)                       // uniform scalar condition
            __builtin_prefetch(ap + k0 + 32, 0, 1);
        v8h a0 = *(const v8h*)(ap + k0);
        v8h a1 = *(const v8h*)(ap + k0 + 16);
        v8h b0 = *(const v8h*)(bp + k0);
        v8h b1 = *(const v8h*)(bp + k0 + 16);
        v16h av = __builtin_shufflevector(a0, a1, 0,1,2,3,4,5,6,7,8,9,10,11,12,13,14,15);
        v16h bv = __builtin_shufflevector(b0, b1, 0,1,2,3,4,5,6,7,8,9,10,11,12,13,14,15);
        acc = __builtin_amdgcn_wmma_f32_16x16x32_f16(false, av, false, bv,
                                                     (short)0, acc, false, false);
    }
#pragma unroll
    for (int r = 0; r < 8; ++r) {
        const int m = tM + grp * 8 + r;
        const int n = tN + lr;
        if (n < N) {
            float v = acc[r] + (bias ? bias[n] : 0.f);
            C[(size_t)m * ldc + n] = v;
            if (C16) C16[(size_t)m * ldc + n] = (h16)v;
        }
    }
}

// ---------------------------------------------------------------------------
// Gather padded f16 A-matrices for m2 / m3 from h (1536 x 128), lda = 32.
// ---------------------------------------------------------------------------
__global__ __launch_bounds__(256)
void k_gather_A2(const float* __restrict__ hbuf, h16* __restrict__ A2) {
    int row = blockIdx.x * 256 + threadIdx.x;      // 32768 rows = (b, w, c)
    if (row >= 32768) return;
    int b = row / 1024, rem = row % 1024, w = rem / 128, c = rem % 128;
    for (int hh = 0; hh < 32; ++hh) {
        float v = (hh < 6) ? hbuf[((size_t)b * 48 + hh * 8 + w) * 128 + c] : 0.f;
        A2[(size_t)row * 32 + hh] = (h16)v;
    }
}
__global__ __launch_bounds__(256)
void k_gather_A3(const float* __restrict__ hbuf, h16* __restrict__ A3) {
    int row = blockIdx.x * 256 + threadIdx.x;      // 24576 rows = (b, c, hh)
    if (row >= 24576) return;
    int b = row / 768, rem = row % 768, c = rem / 6, hh = rem % 6;
    for (int w = 0; w < 32; ++w) {
        float v = (w < 8) ? hbuf[((size_t)b * 48 + hh * 8 + w) * 128 + c] : 0.f;
        A3[(size_t)row * 32 + w] = (h16)v;
    }
}

// ---------------------------------------------------------------------------
// Depthwise 3x3 SAME conv + bias + silu.  Input xz rows (b*L + l), cols [0,D).
// Outputs: xconv (b,d,l) f32 ; a_hw/a_wh f16 rows (b*L+l) with lda=Dpad,
// zero-filled for d in [D, Dpad).
// ---------------------------------------------------------------------------
__global__ __launch_bounds__(256)
void k_conv_silu(const float* __restrict__ xz, int ld,
                 const float* __restrict__ cw, const float* __restrict__ cb,
                 float* __restrict__ xconv, h16* __restrict__ a_hw,
                 h16* __restrict__ a_wh, int Bn, int D, int Dpad, int Hs, int Ws) {
    const int L = Hs * Ws;
    int idx = blockIdx.x * 256 + threadIdx.x;
    if (idx >= Bn * Dpad * L) return;
    int l = idx % L, t = idx / L, d = t % Dpad, b = t / Dpad;
    int h = l / Ws, w = l % Ws;
    int lwh = w * Hs + h;
    if (d >= D) {                                   // zero padding columns
        a_hw[((size_t)b * L + l)   * Dpad + d] = (h16)0.f;
        a_wh[((size_t)b * L + lwh) * Dpad + d] = (h16)0.f;
        return;
    }
    float acc = cb[d];
    for (int dh = 0; dh < 3; ++dh)
        for (int dw = 0; dw < 3; ++dw) {
            int nh = h + dh - 1, nw = w + dw - 1;
            if (nh >= 0 && nh < Hs && nw >= 0 && nw < Ws)
                acc += xz[((size_t)b * L + nh * Ws + nw) * ld + d] * cw[d * 9 + dh * 3 + dw];
        }
    float sv = siluf(acc);
    xconv[((size_t)b * D + d) * L + l] = sv;
    a_hw[((size_t)b * L + l)   * Dpad + d] = (h16)sv;
    a_wh[((size_t)b * L + lwh) * Dpad + d] = (h16)sv;
}

// ---------------------------------------------------------------------------
// delta = softplus(dts @ dt_w^T + dt_bias); dts = first R cols of dbl[k].
// ---------------------------------------------------------------------------
__global__ __launch_bounds__(256)
void k_delta(const float* __restrict__ dbl, const float* __restrict__ dtw,
             const float* __restrict__ dtb, float* __restrict__ delta,
             int Bn, int D, int L, int R, int Ck) {
    int total = 4 * Bn * D * L;
    int idx = blockIdx.x * 256 + threadIdx.x;
    if (idx >= total) return;
    int l = idx % L, t = idx / L, d = t % D; t /= D;
    int b = t % Bn, k = t / Bn;
    const float* row = dbl + ((size_t)k * Bn * L + (size_t)b * L + l) * Ck;
    float acc = dtb[k * D + d];
    for (int r = 0; r < R; ++r) acc += row[r] * dtw[((size_t)(k * D + d)) * R + r];
    delta[(((size_t)k * Bn + b) * D + d) * L + l] = softplusf(acc);
}

// ---------------------------------------------------------------------------
// Selective scan with fused C-contraction. One thread per (k,b,d) row, 16
// states + 16 A-coefs in VGPRs; B/C vectors broadcast through LDS per step.
// Direction flips (k>=2) and H/W transpose (odd k) handled by index remap.
// ---------------------------------------------------------------------------
__global__ __launch_bounds__(64)
void k_scan(const float* __restrict__ dbl, const float* __restrict__ delta,
            const float* __restrict__ xconv, const float* __restrict__ alogs,
            const float* __restrict__ Ds, float* __restrict__ y4,
            int Bn, int D, int L, int Hs, int Ws, int R, int Ck) {
    __shared__ float sB[16], sC[16];
    const int chunks = (D + 63) / 64;
    int bid = blockIdx.x;
    const int chunk = bid % chunks; bid /= chunks;
    const int b = bid % Bn, k = bid / Bn;
    const int d = chunk * 64 + threadIdx.x;
    const bool act = d < D;
    float hst[16], An[16], Dv = 0.f;
#pragma unroll
    for (int n = 0; n < 16; ++n) { hst[n] = 0.f; An[n] = 0.f; }
    if (act) {
#pragma unroll
        for (int n = 0; n < 16; ++n)
            An[n] = -__expf(alogs[((size_t)(k * D + d)) * 16 + n]);
        Dv = Ds[k * D + d];
    }
    const float* dblk = dbl + ((size_t)k * Bn * L + (size_t)b * L) * Ck;
    const float* dl   = delta + (((size_t)k * Bn + b) * D) * L;
    const float* xc   = xconv + (size_t)b * D * L;
    float* yo         = y4 + (((size_t)k * Bn + b) * D) * L;
    const bool flip = (k & 2) != 0, tr = (k & 1) != 0;
    for (int t = 0; t < L; ++t) {
        const int j = flip ? (L - 1 - t) : t;
        __syncthreads();
        if (threadIdx.x < 32) {
            float v = dblk[(size_t)j * Ck + R + threadIdx.x];
            if (threadIdx.x < 16) sB[threadIdx.x] = v;
            else                  sC[threadIdx.x - 16] = v;
        }
        __syncthreads();
        if (act) {
            const int lhw = tr ? ((j % Hs) * Ws + (j / Hs)) : j;
            const float dv = dl[(size_t)d * L + j];
            const float xv = xc[(size_t)d * L + lhw];
            const float dx = dv * xv;
            float y = 0.f;
#pragma unroll
            for (int n = 0; n < 16; ++n) {
                float dA = __expf(dv * An[n]);
                hst[n] = hst[n] * dA + dx * sB[n];
                y += hst[n] * sC[n];
            }
            yo[(size_t)d * L + lhw] = y + Dv * xv;
        }
    }
}

// ---------------------------------------------------------------------------
// Sum 4 scan directions, out_norm LayerNorm over D, gate with silu(z).
// yA is f16 (Mrows x Dpad), zero-padded columns.
// ---------------------------------------------------------------------------
__global__ __launch_bounds__(256)
void k_merge_gate(const float* __restrict__ y4, const float* __restrict__ xz, int ld2,
                  const float* __restrict__ g, const float* __restrict__ be,
                  h16* __restrict__ yA, int Bn, int D, int Dpad, int L) {
    __shared__ float s1[256], s2[256];
    const int row = blockIdx.x;
    const int b = row / L, l = row % L;
    const int d = threadIdx.x;
    float y = 0.f;
    if (d < D)
        for (int k = 0; k < 4; ++k)
            y += y4[(((size_t)k * Bn + b) * D + d) * L + l];
    s1[d] = (d < D) ? y : 0.f;
    s2[d] = (d < D) ? y * y : 0.f;
    __syncthreads();
    for (int off = 128; off > 0; off >>= 1) {
        if (d < off) { s1[d] += s1[d + off]; s2[d] += s2[d + off]; }
        __syncthreads();
    }
    const float mean = s1[0] / (float)D;
    const float var  = s2[0] / (float)D - mean * mean;
    const float inv  = rsqrtf(var + 1e-5f);
    if (d < D) {
        float yn = (y - mean) * inv * g[d] + be[d];
        float z  = xz[(size_t)row * ld2 + D + d];
        yA[(size_t)row * Dpad + d] = (h16)(yn * siluf(z));
    } else if (d < Dpad) {
        yA[(size_t)row * Dpad + d] = (h16)0.f;
    }
}

// ---------------------------------------------------------------------------
// Combine three SS2D branch outputs (inverse permutations), gate with silu(h).
// ---------------------------------------------------------------------------
__global__ __launch_bounds__(256)
void k_combine(const float* __restrict__ hbuf, const float* __restrict__ mo1,
               const float* __restrict__ mo2, const float* __restrict__ mo3,
               h16* __restrict__ gA) {
    int idx = blockIdx.x * 256 + threadIdx.x;
    if (idx >= 1536 * 128) return;
    const int c = idx % 128, row = idx / 128;
    const int b = row / 48, l = row % 48, hh = l / 8, w = l % 8;
    const float hv = hbuf[idx];
    const float v = mo1[idx]
                  + mo2[((size_t)b * 1024 + w * 128 + c) * 6 + hh]
                  + mo3[((size_t)b * 768 + c * 6 + hh) * 8 + w];
    gA[idx] = (h16)(siluf(hv) * v);
}

__global__ void k_zero(float* p, int n) {
    int i = blockIdx.x * 256 + threadIdx.x;
    if (i < n) p[i] = 0.f;
}

// pre += skip*scale (in place), accumulate per-channel BN stats.
__global__ __launch_bounds__(256)
void k_skip_stats(float* __restrict__ pre, const float* __restrict__ x,
                  const float* __restrict__ scale, float* __restrict__ stats) {
    int idx = blockIdx.x * 256 + threadIdx.x;
    if (idx >= 1536 * 1024) return;
    const int dim = idx % 1024, row = idx / 1024;
    const int b = row / 48, l = row % 48;
    float v = pre[idx] + x[(size_t)b * 49152 + (size_t)dim * 48 + l] * scale[0];
    pre[idx] = v;
    atomicAdd(stats + dim, v);
    atomicAdd(stats + 1024 + dim, v * v);
}

__global__ __launch_bounds__(256)
void k_bn_relu(const float* __restrict__ pre, const float* __restrict__ stats,
               const float* __restrict__ g, const float* __restrict__ be,
               float* __restrict__ out) {
    int idx = blockIdx.x * 256 + threadIdx.x;
    if (idx >= 1536 * 1024) return;
    const int dim = idx % 1024, row = idx / 1024;
    const int b = row / 48, l = row % 48;
    const float mean = stats[dim] * (1.f / 1536.f);
    const float var  = stats[1024 + dim] * (1.f / 1536.f) - mean * mean;
    float o = (pre[idx] - mean) * rsqrtf(var + 1e-5f) * g[dim] + be[dim];
    out[(size_t)b * 49152 + (size_t)dim * 48 + l] = fmaxf(o, 0.f);
}

// ---------------------------------------------------------------------------
// Host-side SS2D driver (all launches on `stream`; scratch = bump region).
// ---------------------------------------------------------------------------
struct SS2D {
    const float *A_logs, *Ds, *conv_b, *conv_w, *dt_bias, *dt_w, *in_proj_w,
                *out_norm_b, *out_norm_g, *out_proj_w, *x_proj_w;
};

static void run_ss2d(hipStream_t st, int Bn, int Hs, int Ws, int Dm, int Din, int R,
                     const SS2D& p, const h16* Ain, char* scratch, float* mo) {
    const int L = Hs * Ws, Mrows = Bn * L, Ck = R + 32, D2 = 2 * Din;
    const int Dpad = round32(Din), Kdm = round32(Dm);
    const int NpadIn = round16(D2), NpadX = round16(Ck), NpadOut = round16(Dm);
    char* cur = scratch;
    auto alloc = [&](size_t bytes) -> char* {
        char* r = cur; cur += (bytes + 255) & ~(size_t)255; return r;
    };
    float* xz    = (float*)alloc((size_t)Mrows * D2 * 4);
    float* xconv = (float*)alloc((size_t)Bn * Din * L * 4);
    h16*   a_hw  = (h16*)  alloc((size_t)Mrows * Dpad * 2);
    h16*   a_wh  = (h16*)  alloc((size_t)Mrows * Dpad * 2);
    float* dbl   = (float*)alloc((size_t)4 * Mrows * Ck * 4);
    float* delta = (float*)alloc((size_t)4 * Bn * Din * L * 4);
    float* y4    = (float*)alloc((size_t)4 * Bn * Din * L * 4);
    h16*   yA    = (h16*)  alloc((size_t)Mrows * Dpad * 2);
    h16*   w_in  = (h16*)  alloc((size_t)NpadIn * Kdm * 2);
    h16*   w_xp  = (h16*)  alloc((size_t)4 * NpadX * Dpad * 2);
    h16*   w_out = (h16*)  alloc((size_t)NpadOut * Dpad * 2);

    // weight conversions (f32 -> padded f16)
    k_w2h<<<(NpadIn * Kdm + 255) / 256, 256, 0, st>>>(p.in_proj_w, w_in, D2, Dm, NpadIn, Kdm);
    for (int k = 0; k < 4; ++k)
        k_w2h<<<(NpadX * Dpad + 255) / 256, 256, 0, st>>>(
            p.x_proj_w + (size_t)k * Ck * Din, w_xp + (size_t)k * NpadX * Dpad,
            Ck, Din, NpadX, Dpad);
    k_w2h<<<(NpadOut * Dpad + 255) / 256, 256, 0, st>>>(p.out_proj_w, w_out, Dm, Din,
                                                        NpadOut, Dpad);

    // in_proj
    k_gemm<<<dim3(Mrows / 16, NpadIn / 16), 32, 0, st>>>(
        Ain, Kdm, w_in, Kdm, nullptr, xz, D2, nullptr, D2, Kdm);
    // depthwise conv + silu + direction layouts
    const int nconv = Bn * Dpad * L;
    k_conv_silu<<<(nconv + 255) / 256, 256, 0, st>>>(xz, D2, p.conv_w, p.conv_b,
                                                     xconv, a_hw, a_wh, Bn, Din, Dpad, Hs, Ws);
    // x_proj (4 directions)
    for (int k = 0; k < 4; ++k)
        k_gemm<<<dim3(Mrows / 16, NpadX / 16), 32, 0, st>>>(
            (k & 1) ? a_wh : a_hw, Dpad, w_xp + (size_t)k * NpadX * Dpad, Dpad,
            nullptr, dbl + (size_t)k * Mrows * Ck, Ck, nullptr, Ck, Dpad);
    // dt projection + softplus
    const int nd = 4 * Bn * Din * L;
    k_delta<<<(nd + 255) / 256, 256, 0, st>>>(dbl, p.dt_w, p.dt_bias, delta,
                                              Bn, Din, L, R, Ck);
    // selective scan (fused C contraction)
    const int chunks = (Din + 63) / 64;
    k_scan<<<4 * Bn * chunks, 64, 0, st>>>(dbl, delta, xconv, p.A_logs, p.Ds, y4,
                                           Bn, Din, L, Hs, Ws, R, Ck);
    // merge directions + out_norm + silu(z) gate
    k_merge_gate<<<Mrows, 256, 0, st>>>(y4, xz, D2, p.out_norm_g, p.out_norm_b,
                                        yA, Bn, Din, Dpad, L);
    // out_proj
    k_gemm<<<dim3(Mrows / 16, NpadOut / 16), 32, 0, st>>>(
        yA, Dpad, w_out, Dpad, nullptr, mo, Dm, nullptr, Dm, Dpad);
}

// ---------------------------------------------------------------------------
// Input order: jax pytree (recursively sorted-key) flattening of
// {'params': {...}, 'x': ...}  ->  params leaves (sorted) then x.
// ---------------------------------------------------------------------------
extern "C" void kernel_launch(void* const* d_in, const int* in_sizes, int n_in,
                              void* d_out, int out_size, void* d_ws, size_t ws_size,
                              hipStream_t stream) {
    (void)in_sizes; (void)n_in; (void)out_size; (void)ws_size;
    const float* bn_b = (const float*)d_in[0];
    const float* bn_g = (const float*)d_in[1];
    const float* ln_b = (const float*)d_in[2];
    const float* ln_g = (const float*)d_in[3];
    auto mod = [&](int base) -> SS2D {
        SS2D m;
        m.A_logs     = (const float*)d_in[base + 0];
        m.Ds         = (const float*)d_in[base + 1];
        m.conv_b     = (const float*)d_in[base + 2];
        m.conv_w     = (const float*)d_in[base + 3];
        m.dt_bias    = (const float*)d_in[base + 4];
        m.dt_w       = (const float*)d_in[base + 5];
        m.in_proj_w  = (const float*)d_in[base + 6];
        m.out_norm_b = (const float*)d_in[base + 7];
        m.out_norm_g = (const float*)d_in[base + 8];
        m.out_proj_w = (const float*)d_in[base + 9];
        m.x_proj_w   = (const float*)d_in[base + 10];
        return m;
    };
    SS2D m1 = mod(4), m2 = mod(15), m3 = mod(26);
    const float* proj_in_b  = (const float*)d_in[37];
    const float* proj_in_w  = (const float*)d_in[38];
    const float* proj_out_b = (const float*)d_in[39];
    const float* proj_out_w = (const float*)d_in[40];
    const float* scale      = (const float*)d_in[41];
    const float* x          = (const float*)d_in[42];

    char* cur = (char*)d_ws;
    auto alloc = [&](size_t bytes) -> char* {
        char* r = cur; cur += (bytes + 255) & ~(size_t)255; return r;
    };
    h16*   hln    = (h16*)  alloc((size_t)1536 * 1024 * 2);
    float* hbuf   = (float*)alloc((size_t)1536 * 128 * 4);
    h16*   hhalf  = (h16*)  alloc((size_t)1536 * 128 * 2);
    h16*   A2     = (h16*)  alloc((size_t)32768 * 32 * 2);
    h16*   A3     = (h16*)  alloc((size_t)24576 * 32 * 2);
    float* mo1    = (float*)alloc((size_t)1536 * 128 * 4);
    float* mo2    = (float*)alloc((size_t)32768 * 6 * 4);
    float* mo3    = (float*)alloc((size_t)24576 * 8 * 4);
    h16*   gA     = (h16*)  alloc((size_t)1536 * 128 * 2);
    float* pre    = (float*)alloc((size_t)1536 * 1024 * 4);
    float* stats  = (float*)alloc(2048 * 4);
    h16*   pin16  = (h16*)  alloc((size_t)128 * 1024 * 2);
    h16*   pout16 = (h16*)  alloc((size_t)1024 * 128 * 2);
    char*  mscr   = alloc((size_t)56 * 1024 * 1024);   // per-module scratch (reused)

    // weight conversions for the two big projections
    k_w2h<<<(128 * 1024 + 255) / 256, 256, 0, stream>>>(proj_in_w, pin16, 128, 1024, 128, 1024);
    k_w2h<<<(1024 * 128 + 255) / 256, 256, 0, stream>>>(proj_out_w, pout16, 1024, 128, 1024, 128);

    // 1) LayerNorm over DIM -> f16 A (1536 x 1024)
    k_ln_in<<<1536, 256, 0, stream>>>(x, ln_g, ln_b, hln);

    // 2) proj_in GEMM: (1536x1024)x(128x1024)^T -> h (f32 + f16 mirror)
    k_gemm<<<dim3(1536 / 16, 128 / 16), 32, 0, stream>>>(
        hln, 1024, pin16, 1024, proj_in_b, hbuf, 128, hhalf, 128, 1024);

    // 3) gather permuted A matrices for m2/m3 (lda = 32, zero padded)
    k_gather_A2<<<(32768 + 255) / 256, 256, 0, stream>>>(hbuf, A2);
    k_gather_A3<<<(24576 + 255) / 256, 256, 0, stream>>>(hbuf, A3);

    // 4) three SS2D branches (sequential, reusing scratch)
    run_ss2d(stream, 32, 6,   8,  128, 256, 8, m1, hhalf, mscr, mo1);
    run_ss2d(stream, 32, 8, 128,    6,  12, 1, m2, A2,    mscr, mo2);
    run_ss2d(stream, 32, 128,  6,   8,  16, 1, m3, A3,    mscr, mo3);

    // 5) silu(h) * (x1+x2+x3) -> f16 (1536 x 128)
    k_combine<<<(1536 * 128 + 255) / 256, 256, 0, stream>>>(hbuf, mo1, mo2, mo3, gA);

    // 6) proj_out GEMM: (1536x128)x(1024x128)^T + bias -> pre
    k_gemm<<<dim3(1536 / 16, 1024 / 16), 32, 0, stream>>>(
        gA, 128, pout16, 128, proj_out_b, pre, 1024, nullptr, 1024, 128);

    // 7) skip*scale + BN stats + BN/relu -> d_out (B,DIM,H,W)
    k_zero<<<8, 256, 0, stream>>>(stats, 2048);
    k_skip_stats<<<(1536 * 1024 + 255) / 256, 256, 0, stream>>>(pre, x, scale, stats);
    k_bn_relu<<<(1536 * 1024 + 255) / 256, 256, 0, stream>>>(pre, stats, bn_g, bn_b,
                                                             (float*)d_out);
}
